// RNNModel_67783173865596
// MI455X (gfx1250) — matrix operand order repaired and checked by reading, not compiled, not run on hardware
//
#include <hip/hip_runtime.h>
#include <stdint.h>

#define DEVINL __device__ __forceinline__

typedef __attribute__((ext_vector_type(16))) __bf16   v16bf;
typedef __attribute__((ext_vector_type(8)))  float    v8f;
typedef __attribute__((ext_vector_type(4)))  uint32_t u32x4;

static constexpr int B_  = 64;
static constexpr int T_  = 2048;      // power of two: m = b*T + t -> b = m>>11, t = m&2047
static constexpr int IN_ = 256;
static constexpr int H_  = 512;
static constexpr int C_  = 128;
static constexpr int M_  = B_ * T_;   // 131072 flattened rows
static constexpr int NSYNC_WG = 16;   // workgroups in the scan kernel

// ---------------- bf16 helpers ----------------
DEVINL __bf16 f2bf(float f) {
  uint32_t u = __builtin_bit_cast(uint32_t, f);
  uint32_t r = u + 0x7FFFu + ((u >> 16) & 1u);   // round-to-nearest-even
  uint16_t h = (uint16_t)(r >> 16);
  return __builtin_bit_cast(__bf16, h);
}
DEVINL float bf2f(__bf16 b) {
  uint32_t u = ((uint32_t)__builtin_bit_cast(uint16_t, b)) << 16;
  return __builtin_bit_cast(float, u);
}

union BF16x16 { v16bf v; u32x4 q[2]; };

// A fragment: 16x32 (MxK) bf16, row-major source with leading dim ld.
// lane l (r=l&15, hi=l>>4): elems 0-7 = A[m0+r][k0+8*hi .. +7],
//                           elems 8-15 = A[m0+r][k0+16+8*hi .. +7]
DEVINL v16bf load_frag_a(const __bf16* base, int ld, int m0, int k0, int lane) {
  const int r = lane & 15, hi = lane >> 4;
  const __bf16* p = base + (size_t)(m0 + r) * ld + k0 + 8 * hi;
  BF16x16 u;
  u.q[0] = *(const u32x4*)(p);
  u.q[1] = *(const u32x4*)(p + 16);
  return u.v;
}
// B fragment: 32x16 (KxN) where B[k][n] = W[n][k], W row-major [N x K], leading dim ld.
// lane l: n = n0 + (l&15); elems 0-7 = W[n][k0+8*hi .. +7], elems 8-15 = W[n][k0+16+8*hi .. +7]
DEVINL v16bf load_frag_b(const __bf16* w, int ld, int n0, int k0, int lane) {
  const int r = lane & 15, hi = lane >> 4;
  const __bf16* p = w + (size_t)(n0 + r) * ld + k0 + 8 * hi;
  BF16x16 u;
  u.q[0] = *(const u32x4*)(p);
  u.q[1] = *(const u32x4*)(p + 16);
  return u.v;
}
DEVINL v8f wmma_bf16(v16bf a, v16bf b, v8f c) {
  return __builtin_amdgcn_wmma_f32_16x16x32_bf16(false, a, false, b, (short)0, c,
                                                 false, false);
}

// ---------------- kernels ----------------
__global__ void k_cvt_bf16(const float* __restrict__ in, __bf16* __restrict__ out, int n) {
  int i = blockIdx.x * blockDim.x + threadIdx.x;
  if (i < n) out[i] = f2bf(in[i]);
}

// out = A[M,K] @ W^T (+ b0 + b1), bf16 out.
// STEP_MAJOR=0: out row = m (row-major [M, 512])
// STEP_MAJOR=1: out row = t*B + b where m = b*T + t  (layout [T][64][512])
template <int STEP_MAJOR>
__global__ __launch_bounds__(256) void k_gemm_proj(
    const __bf16* __restrict__ A, const __bf16* __restrict__ W,
    const float* __restrict__ b0, const float* __restrict__ b1,
    __bf16* __restrict__ out, int K) {
  const int lane = threadIdx.x & 31;
  const int wave = threadIdx.x >> 5;
  const int m0 = blockIdx.x * 128 + wave * 16;
  const int n0 = blockIdx.y * 64;
  const v8f zero = {0.f,0.f,0.f,0.f,0.f,0.f,0.f,0.f};
  v8f acc[4] = {zero, zero, zero, zero};
  for (int k0 = 0; k0 < K; k0 += 32) {
    v16bf a = load_frag_a(A, K, m0, k0, lane);
#pragma unroll
    for (int nb = 0; nb < 4; ++nb) {
      v16bf b = load_frag_b(W, K, n0 + nb * 16, k0, lane);
      acc[nb] = wmma_bf16(a, b, acc[nb]);
    }
  }
  const int r = lane & 15, hi = lane >> 4;
#pragma unroll
  for (int nb = 0; nb < 4; ++nb) {
    const int n = n0 + nb * 16 + r;
    const float bias = b0[n] + b1[n];
#pragma unroll
    for (int e = 0; e < 8; ++e) {
      const int m = m0 + e + 8 * hi;
      size_t row;
      if (STEP_MAJOR) {
        const int b = m >> 11;          // m / T
        const int t = m & (T_ - 1);     // m % T
        row = (size_t)t * B_ + b;
      } else {
        row = (size_t)m;
      }
      out[row * H_ + n] = f2bf(acc[nb][e] + bias);
    }
  }
}

// Recurrent scan: h_t = relu(pre[t,:,:] + h_{t-1} @ Whh^T).
// 16 WGs, each owns 32 output columns; the W_hh slice lives in registers as
// pre-packed WMMA B fragments for the whole scan (zero weight traffic in the
// loop). Triple-buffered h state; one device-scope arrive barrier per step.
// pre is step-major [T][64][512]; next step's 64KB slice is prefetched.
template <int STORE_SEQ>
__global__ __launch_bounds__(256, 1) void k_rnn_scan(
    const __bf16* __restrict__ pre,   // [T][64][512] bf16 (step-major)
    const __bf16* __restrict__ Whh,   // [512, 512] bf16
    __bf16* __restrict__ hbuf,        // [3][64][512] bf16, zeroed
    __bf16* __restrict__ hseq,        // [B*T, 512] bf16 (row = b*T + t) or unused
    int* __restrict__ syncc) {        // zeroed
  const int lane = threadIdx.x & 31;
  const int wave = threadIdx.x >> 5;
  const int m0 = (wave & 3) * 16;                    // batch tile
  const int n0 = blockIdx.x * 32 + (wave >> 2) * 16; // output-column tile

  // Persistent register-resident weight fragments (16 x v16bf = 128 VGPRs).
  v16bf bfrag[16];
#pragma unroll
  for (int kc = 0; kc < 16; ++kc)
    bfrag[kc] = load_frag_b(Whh, H_, n0, kc * 32, lane);

  const int r = lane & 15, hi = lane >> 4;
  const v8f zero = {0.f,0.f,0.f,0.f,0.f,0.f,0.f,0.f};

  // This WG's slice of the per-step pre tile (for prefetch addressing):
  // 64 rows x 32 cols; thread i covers row i/4, 8 elements at col (i%4)*8.
  const int pf_row = threadIdx.x >> 2;
  const int pf_col = blockIdx.x * 32 + (threadIdx.x & 3) * 8;

  for (int t = 0; t < T_; ++t) {
    const __bf16* hin     = hbuf + (size_t)(t % 3) * (B_ * H_);
    __bf16*       hout    = hbuf + (size_t)((t + 1) % 3) * (B_ * H_);
    const __bf16* pre_t   = pre + (size_t)t * (B_ * H_);

    // Stream next step's pre slice toward the caches while we compute.
    if (t + 1 < T_) {
      const __bf16* pre_n = pre + (size_t)(t + 1) * (B_ * H_);
      __builtin_prefetch(pre_n + (size_t)pf_row * H_ + pf_col, 0, 1);
    }

    // Two independent accumulator chains so consecutive WMMAs pipeline.
    v8f acc0 = zero, acc1 = zero;
#pragma unroll
    for (int kc = 0; kc < 16; kc += 2) {
      v16bf a0 = load_frag_a(hin, H_, m0, kc * 32, lane);
      v16bf a1 = load_frag_a(hin, H_, m0, (kc + 1) * 32, lane);
      acc0 = wmma_bf16(a0, bfrag[kc], acc0);
      acc1 = wmma_bf16(a1, bfrag[kc + 1], acc1);
    }

#pragma unroll
    for (int e = 0; e < 8; ++e) {
      const int b = m0 + e + 8 * hi;                 // batch index
      const int n = n0 + r;
      float v = acc0[e] + acc1[e] + bf2f(pre_t[b * H_ + n]);
      v = v > 0.f ? v : 0.f;
      const __bf16 hv = f2bf(v);
      hout[b * H_ + n] = hv;
      if (STORE_SEQ) hseq[((size_t)b * T_ + t) * H_ + n] = hv;
    }

    // Device-scope step barrier across the 16 WGs.
    __threadfence();
    __syncthreads();
    if (threadIdx.x == 0) {
      __hip_atomic_fetch_add(syncc, 1, __ATOMIC_RELEASE, __HIP_MEMORY_SCOPE_AGENT);
      const int target = NSYNC_WG * (t + 1);
      while (__hip_atomic_load(syncc, __ATOMIC_ACQUIRE, __HIP_MEMORY_SCOPE_AGENT) < target)
        __builtin_amdgcn_s_sleep(2);
    }
    __syncthreads();
  }
}

// out[64,128] = h_last[64,512] @ fc_w^T + fc_b   (f32 out)
__global__ __launch_bounds__(256) void k_fc(
    const __bf16* __restrict__ hlast, const __bf16* __restrict__ W,
    const float* __restrict__ bias, float* __restrict__ out) {
  const int lane = threadIdx.x & 31;
  const int wave = threadIdx.x >> 5;
  const int m0 = (wave & 3) * 16;
  const int nhalf = (wave >> 2) * 64;
  const v8f zero = {0.f,0.f,0.f,0.f,0.f,0.f,0.f,0.f};
  v8f acc[4] = {zero, zero, zero, zero};
  for (int k0 = 0; k0 < H_; k0 += 32) {
    v16bf a = load_frag_a(hlast, H_, m0, k0, lane);
#pragma unroll
    for (int nb = 0; nb < 4; ++nb) {
      v16bf b = load_frag_b(W, H_, nhalf + nb * 16, k0, lane);
      acc[nb] = wmma_bf16(a, b, acc[nb]);
    }
  }
  const int r = lane & 15, hi = lane >> 4;
#pragma unroll
  for (int nb = 0; nb < 4; ++nb) {
    const int n = nhalf + nb * 16 + r;
    const float bv = bias[n];
#pragma unroll
    for (int e = 0; e < 8; ++e) {
      const int m = m0 + e + 8 * hi;
      out[(size_t)m * C_ + n] = acc[nb][e] + bv;
    }
  }
}

// ---------------- workspace layout (bytes) ----------------
static constexpr size_t OFF_XBF  = 0;                                    // x in bf16
static constexpr size_t OFF_WIH0 = OFF_XBF  + (size_t)M_ * IN_ * 2;
static constexpr size_t OFF_WHH0 = OFF_WIH0 + (size_t)H_ * IN_ * 2;
static constexpr size_t OFF_WIH1 = OFF_WHH0 + (size_t)H_ * H_ * 2;
static constexpr size_t OFF_WHH1 = OFF_WIH1 + (size_t)H_ * H_ * 2;
static constexpr size_t OFF_FCW  = OFF_WHH1 + (size_t)H_ * H_ * 2;
static constexpr size_t OFF_PRE  = OFF_FCW  + (size_t)C_ * H_ * 2;       // [T][64][512] bf16
static constexpr size_t OFF_HSEQ = OFF_PRE  + (size_t)M_ * H_ * 2;       // [B*T][512] bf16
static constexpr size_t OFF_HBUF = OFF_HSEQ + (size_t)M_ * H_ * 2;       // [3][64][512] bf16
static constexpr size_t OFF_SYNC = OFF_HBUF + (size_t)3 * B_ * H_ * 2;
static constexpr size_t ZERO_SZ  = (size_t)3 * B_ * H_ * 2 + 256;

extern "C" void kernel_launch(void* const* d_in, const int* in_sizes, int n_in,
                              void* d_out, int out_size, void* d_ws, size_t ws_size,
                              hipStream_t stream) {
  (void)in_sizes; (void)n_in; (void)out_size; (void)ws_size;
  const float* x     = (const float*)d_in[0];
  const float* W_ih0 = (const float*)d_in[1];
  const float* W_hh0 = (const float*)d_in[2];
  const float* b_ih0 = (const float*)d_in[3];
  const float* b_hh0 = (const float*)d_in[4];
  const float* W_ih1 = (const float*)d_in[5];
  const float* W_hh1 = (const float*)d_in[6];
  const float* b_ih1 = (const float*)d_in[7];
  const float* b_hh1 = (const float*)d_in[8];
  const float* fc_w  = (const float*)d_in[9];
  const float* fc_b  = (const float*)d_in[10];
  float* out = (float*)d_out;

  char* ws = (char*)d_ws;
  __bf16* xbf   = (__bf16*)(ws + OFF_XBF);
  __bf16* wih0  = (__bf16*)(ws + OFF_WIH0);
  __bf16* whh0  = (__bf16*)(ws + OFF_WHH0);
  __bf16* wih1  = (__bf16*)(ws + OFF_WIH1);
  __bf16* whh1  = (__bf16*)(ws + OFF_WHH1);
  __bf16* fcw   = (__bf16*)(ws + OFF_FCW);
  __bf16* pre   = (__bf16*)(ws + OFF_PRE);
  __bf16* hseq  = (__bf16*)(ws + OFF_HSEQ);
  __bf16* hbuf  = (__bf16*)(ws + OFF_HBUF);
  int*    syncc = (int*)   (ws + OFF_SYNC);

  // 1) fp32 -> bf16 conversions
  auto cvt = [&](const float* src, __bf16* dst, int n) {
    k_cvt_bf16<<<dim3((n + 255) / 256), dim3(256), 0, stream>>>(src, dst, n);
  };
  cvt(x,     xbf,  M_ * IN_);
  cvt(W_ih0, wih0, H_ * IN_);
  cvt(W_hh0, whh0, H_ * H_);
  cvt(W_ih1, wih1, H_ * H_);
  cvt(W_hh1, whh1, H_ * H_);
  cvt(fc_w,  fcw,  C_ * H_);

  // 2) layer-0 input projection: pre0[t][b][:] = x[b,t,:] @ W_ih0^T + b_ih0 + b_hh0
  k_gemm_proj<1><<<dim3(M_ / 128, H_ / 64), dim3(256), 0, stream>>>(
      xbf, wih0, b_ih0, b_hh0, pre, IN_);

  // 3) layer-0 scan (stores full hidden sequence in row = b*T+t layout)
  hipMemsetAsync(ws + OFF_HBUF, 0, ZERO_SZ, stream);
  k_rnn_scan<1><<<dim3(NSYNC_WG), dim3(256), 0, stream>>>(pre, whh0, hbuf, hseq, syncc);

  // 4) layer-1 input projection: pre1 = hseq @ W_ih1^T + b_ih1 + b_hh1 (reuses pre)
  k_gemm_proj<1><<<dim3(M_ / 128, H_ / 64), dim3(256), 0, stream>>>(
      hseq, wih1, b_ih1, b_hh1, pre, H_);

  // 5) layer-1 scan (final h only)
  hipMemsetAsync(ws + OFF_HBUF, 0, ZERO_SZ, stream);
  k_rnn_scan<0><<<dim3(NSYNC_WG), dim3(256), 0, stream>>>(pre, whh1, hbuf, hseq, syncc);

  // 6) FC head from h_{T-1}, which lives in buffer index T%3 == 2
  const __bf16* hlast = hbuf + (size_t)(T_ % 3) * (B_ * H_);
  k_fc<<<dim3(1), dim3(256), 0, stream>>>(hlast, fcw, fc_b, out);
}
